// Einsum_64656437674444
// MI455X (gfx1250) — matrix operand order repaired and checked by reading, not compiled, not run on hardware
//
#include <hip/hip_runtime.h>
#include <hip/hip_fp16.h>
#include <stdint.h>

// out[M,N] = xh[M,K] * w[K,N] ; M=B*T, K=D, N=F
#define M_TOT 4096
#define K_TOT 3072
#define N_TOT 24576

#define BM 128
#define BN 256
#define BK 32
#define LDA_H (BK + 8)    // lA row stride in halfs ([m][k] layout)
#define LDB_H (BN + 8)    // lB row stride in halfs ([k][n] layout)

typedef __attribute__((ext_vector_type(16))) _Float16 v16h;
typedef __attribute__((ext_vector_type(8)))  _Float16 v8h;
typedef __attribute__((ext_vector_type(2)))  _Float16 h2v;
typedef __attribute__((ext_vector_type(8)))  float    v8f;
typedef __attribute__((ext_vector_type(4)))  float    f4;
typedef __attribute__((ext_vector_type(4)))  unsigned int u4;

// ---------------------------------------------------------------------------
// Kernel 1: fold per-K scale into x, convert to f16:  xh[m,k] = x[m,k]*s[k]
// ---------------------------------------------------------------------------
__global__ __launch_bounds__(256) void fold_scale_kernel(
    const float* __restrict__ x,
    const _Float16* __restrict__ s,
    _Float16* __restrict__ xh)
{
  const size_t i0 = ((size_t)blockIdx.x * 256 + threadIdx.x) * 8;
  const int k0 = (int)(i0 % K_TOT);
  f4 x0 = *(const f4*)(x + i0);
  f4 x1 = *(const f4*)(x + i0 + 4);
  v8h sv = *(const v8h*)(s + k0);
  v8h r;
#pragma unroll
  for (int j = 0; j < 4; ++j) r[j]     = (_Float16)(x0[j] * (float)sv[j]);
#pragma unroll
  for (int j = 0; j < 4; ++j) r[4 + j] = (_Float16)(x1[j] * (float)sv[4 + j]);
  *(v8h*)(xh + i0) = r;
}

// ---------------------------------------------------------------------------
// Helpers
// ---------------------------------------------------------------------------
__device__ __forceinline__ void async_copy_a32B(uint32_t ldsOff, const _Float16* g) {
  // 32B per thread via 2x async b128 global->LDS (CDNA5 async path, ASYNCcnt)
  uint64_t ga = (uint64_t)(uintptr_t)g;
  asm volatile("global_load_async_to_lds_b128 %0, %1, off"
               :: "v"(ldsOff), "v"(ga) : "memory");
  asm volatile("global_load_async_to_lds_b128 %0, %1, off"
               :: "v"(ldsOff + 16u), "v"(ga + 16u) : "memory");
}

// Exact int8 -> f16 unpack: u = b^0x80 = b+128; 0x6400|u == 1024+u (exact in
// f16); pk_add -1152 -> b.  5 VALU (xor + 2 perm + 2 pk_add) per 4 weights.
__device__ __forceinline__ uint32_t pair_to_f16(uint32_t xw, uint32_t sel) {
  union { uint32_t u; h2v h; } t;
  t.u = __builtin_amdgcn_perm(0x64646464u, xw, sel);
  const h2v bias = { (_Float16)-1152.0f, (_Float16)-1152.0f };
  t.h = t.h + bias;                       // v_pk_add_f16
  return t.u;
}

__device__ __forceinline__ void dequant16(u4 w, v8h* lo, v8h* hi) {
  union { v8h h; uint32_t u[4]; } a, b;
  uint32_t x0 = w[0] ^ 0x80808080u;
  uint32_t x1 = w[1] ^ 0x80808080u;
  uint32_t x2 = w[2] ^ 0x80808080u;
  uint32_t x3 = w[3] ^ 0x80808080u;
  a.u[0] = pair_to_f16(x0, 0x05010400u);  // bytes 0,1 -> halfs 0,1
  a.u[1] = pair_to_f16(x0, 0x07030602u);  // bytes 2,3 -> halfs 2,3
  a.u[2] = pair_to_f16(x1, 0x05010400u);
  a.u[3] = pair_to_f16(x1, 0x07030602u);
  b.u[0] = pair_to_f16(x2, 0x05010400u);
  b.u[1] = pair_to_f16(x2, 0x07030602u);
  b.u[2] = pair_to_f16(x3, 0x05010400u);
  b.u[3] = pair_to_f16(x3, 0x07030602u);
  *lo = a.h;
  *hi = b.h;
}

__device__ __forceinline__ void dequant_store_b(_Float16* p0, _Float16* p1,
                                                u4 w0, u4 w1) {
  v8h q0, q1, q2, q3;
  dequant16(w0, &q0, &q1);
  dequant16(w1, &q2, &q3);
  *(v8h*)p0       = q0;
  *(v8h*)(p0 + 8) = q1;
  *(v8h*)p1       = q2;
  *(v8h*)(p1 + 8) = q3;
}

// One K-step of wave-level MMA: 4 A-frags x 4 B-frags -> 16 WMMAs.
__device__ __forceinline__ void wave_mma(const _Float16* baseA,
                                         const _Float16* baseB,
                                         int mBase, int nBase,
                                         int lrow, int lsel, int lane,
                                         v8f (&acc)[4][4]) {
  // A fragments: lane = M row; lanes 0-15 K0-7/16-23, lanes 16-31 K8-15/24-31
  v16h afrag[4];
#pragma unroll
  for (int mi = 0; mi < 4; ++mi) {
    const _Float16* p = baseA + (mBase + mi * 16 + lrow) * LDA_H + lsel;
    union { v16h v; v8h h[2]; } u;
    u.h[0] = *(const v8h*)p;
    u.h[1] = *(const v8h*)(p + 16);
    afrag[mi] = u.v;
  }
  // B fragments: lane = N col; lanes 0-15 K0-7/16-23, lanes 16-31 K8-15/24-31
  v16h bfrag[4];
#pragma unroll
  for (int ni = 0; ni < 4; ++ni) {
    const _Float16* p = baseB + lsel * LDB_H + nBase + ni * 16 + lrow;
    v16h v;
#pragma unroll
    for (int s = 0; s < 16; ++s)
      v[s] = p[((s & 7) + (s >> 3) * 16) * LDB_H];
    bfrag[ni] = v;
  }
#pragma unroll
  for (int mi = 0; mi < 4; ++mi)
#pragma unroll
    for (int ni = 0; ni < 4; ++ni)
      acc[mi][ni] = __builtin_amdgcn_wmma_f32_16x16x32_f16(
          false, afrag[mi], false, bfrag[ni],
          (short)0, acc[mi][ni], false, false);
}

// ---------------------------------------------------------------------------
// Kernel 2: WMMA GEMM, double-buffered LDS software pipeline, peeled tail.
// Block tile 128x256, 256 threads = 8 waves (2x4), wave tile 64x64.
// ---------------------------------------------------------------------------
__global__ __launch_bounds__(256) void gemm_wmma_kernel(
    const _Float16* __restrict__ A,   // [M,K] f16 (x*s)
    const char* __restrict__ W,       // [K,N] int8
    float* __restrict__ out)          // [M,N] f32
{
  __shared__ _Float16 lA[2][BM * LDA_H];   // [m][k]
  __shared__ _Float16 lB[2][BK * LDB_H];   // [k][n]

  const int tid  = threadIdx.x;
  const int lane = tid & 31;
  const int wid  = tid >> 5;
  const int blockM = blockIdx.y * BM;
  const int blockN = blockIdx.x * BN;

  const int mBase = (wid >> 2) * 64;   // 0 or 64
  const int nBase = (wid & 3) * 64;    // 0,64,128,192

  v8f acc[4][4];
#pragma unroll
  for (int i = 0; i < 4; ++i)
#pragma unroll
    for (int j = 0; j < 4; ++j)
      acc[i][j] = {};

  // A staging: 128 rows x 32 halfs (8KB); thread -> 32B via 2 async b128.
  const int aRow = tid >> 1;
  const int aOff = (tid & 1) * 16;                 // halfs
  const _Float16* gA = A + (size_t)(blockM + aRow) * K_TOT + aOff;
  // LDS byte offset = flat addr low 32 bits (shared aperture mapping)
  const uint32_t ldsA[2] = {
    (uint32_t)(uintptr_t)&lA[0][aRow * LDA_H + aOff],
    (uint32_t)(uintptr_t)&lA[1][aRow * LDA_H + aOff],
  };

  // W staging: 32 rows x 256 int8; thread -> rows bRow, bRow+16, 16B each.
  const int bRow = tid >> 4;                       // 0..15
  const int bOff = (tid & 15) * 16;                // 0..240
  const char* gW = W + (size_t)bRow * N_TOT + blockN + bOff;
  _Float16* pB0[2] = { &lB[0][bRow * LDB_H + bOff],
                       &lB[1][bRow * LDB_H + bOff] };
  _Float16* pB1[2] = { &lB[0][(bRow + 16) * LDB_H + bOff],
                       &lB[1][(bRow + 16) * LDB_H + bOff] };

  const int lsel = (lane >> 4) * 8;                // A half-select (halfs)
  const int lrow = lane & 15;

  // ---- pipeline prologue: fill buffer 0 ----
  async_copy_a32B(ldsA[0], gA);
  dequant_store_b(pB0[0], pB1[0],
                  *(const u4*)gW,
                  *(const u4*)(gW + (size_t)16 * N_TOT));
  asm volatile("s_wait_asynccnt 0x0" ::: "memory");
  __syncthreads();

  // ---- steady state: 95 branch-free iterations with prefetch ----
  int buf = 0;
  for (int kt = 0; kt < K_TOT - BK; kt += BK, buf ^= 1) {
    const int ktn = kt + BK;
    // prefetch next W tile (registers) + next A tile (async DMA to LDS)
    u4 w0 = *(const u4*)(gW + (size_t)ktn * N_TOT);
    u4 w1 = *(const u4*)(gW + (size_t)(ktn + 16) * N_TOT);
    async_copy_a32B(ldsA[buf ^ 1], gA + ktn);

    wave_mma(&lA[buf][0], &lB[buf][0], mBase, nBase, lrow, lsel, lane, acc);

    // finish filling next buffer while WMMAs drain, then one barrier
    dequant_store_b(pB0[buf ^ 1], pB1[buf ^ 1], w0, w1);
    asm volatile("s_wait_asynccnt 0x0" ::: "memory");
    __syncthreads();
  }

  // ---- tail: compute-only on the last buffer ----
  wave_mma(&lA[buf][0], &lB[buf][0], mBase, nBase, lrow, lsel, lane, acc);

  // Epilogue: C layout — VGPR j: lanes 0-15 -> M=j, lanes 16-31 -> M=j+8
  const int rOff = (lane >> 4) * 8;
  const int cCol = lane & 15;
#pragma unroll
  for (int mi = 0; mi < 4; ++mi) {
#pragma unroll
    for (int ni = 0; ni < 4; ++ni) {
      const int m0 = blockM + mBase + mi * 16 + rOff;
      const int n  = blockN + nBase + ni * 16 + cCol;
#pragma unroll
      for (int j = 0; j < 8; ++j)
        out[(size_t)(m0 + j) * N_TOT + n] = acc[mi][ni][j];
    }
  }
}

// ---------------------------------------------------------------------------
extern "C" void kernel_launch(void* const* d_in, const int* in_sizes, int n_in,
                              void* d_out, int out_size, void* d_ws, size_t ws_size,
                              hipStream_t stream) {
  const float*    x = (const float*)d_in[0];
  const char*     w = (const char*)d_in[1];
  const _Float16* s = (const _Float16*)d_in[2];
  float* out   = (float*)d_out;
  _Float16* xh = (_Float16*)d_ws;    // 25.2 MB f16 scratch

  const int foldBlocks = (M_TOT * K_TOT / 8) / 256;
  fold_scale_kernel<<<foldBlocks, 256, 0, stream>>>(x, s, xh);

  dim3 grid(N_TOT / BN, M_TOT / BM);   // 96 x 32
  gemm_wmma_kernel<<<grid, 256, 0, stream>>>(xh, w, out);
}
